// AttentionHead_25709674234498
// MI455X (gfx1250) — compile-verified, matrix operand order checked by reading, not run on hardware
//
#include <hip/hip_runtime.h>
#include <hip/hip_bf16.h>

typedef __attribute__((ext_vector_type(16))) _Float16 v16h;
typedef __attribute__((ext_vector_type(8)))  float    v8f;
typedef __attribute__((ext_vector_type(4)))  int      v4i;
typedef __attribute__((ext_vector_type(8)))  int      v8i;
typedef __attribute__((ext_vector_type(4)))  unsigned v4u;

#define B_  8
#define S_  2048
#define H_  1024
#define D_  64

#define AS1 __attribute__((address_space(1)))
#define AS3 __attribute__((address_space(3)))

union Frag {
  v16h v;
  _Float16 h[16];
  unsigned u[8];
};

// 16-bit A/B operand layout (ISA 7.12.2): per lane, the 16 halves are two
// contiguous 8-half runs: [8*half, 8*half+8) and [16+8*half, 16+8*half+8).
// So any fragment is exactly two b128 loads from a contiguous 32-half row.
__device__ __forceinline__ void load_frag_h16(Frag& f, const _Float16* base,
                                              int half) {
  const uint4 lo = *(const uint4*)(base + 8 * half);
  const uint4 hi = *(const uint4*)(base + 16 + 8 * half);
  f.u[0] = lo.x; f.u[1] = lo.y; f.u[2] = lo.z; f.u[3] = lo.w;
  f.u[4] = hi.x; f.u[5] = hi.y; f.u[6] = hi.z; f.u[7] = hi.w;
}

// ---- CDNA5 async global->LDS copy (ASYNCcnt) ------------------------------
#if __has_builtin(__builtin_amdgcn_global_load_async_to_lds_b128)
#define HAVE_ASYNC_LDS 1
#endif

__device__ __forceinline__ void cp_async_b128(void* lds, const void* g) {
#if defined(HAVE_ASYNC_LDS)
  __builtin_amdgcn_global_load_async_to_lds_b128(
      (AS1 v4i*)(v4i*)g, (AS3 v4i*)(v4i*)lds, 0, 0);
#else
  *(uint4*)lds = *(const uint4*)g;   // synchronous fallback
#endif
}

template <int N>
__device__ __forceinline__ void wait_async() {
#if __has_builtin(__builtin_amdgcn_s_wait_asynccnt)
  __builtin_amdgcn_s_wait_asynccnt(N);
#else
  asm volatile("s_wait_asynccnt %0" ::"i"(N) : "memory");
#endif
}

// ---- CDNA5 Tensor Data Mover (TENSORcnt) ----------------------------------
#if __has_builtin(__builtin_amdgcn_tensor_load_to_lds)
#define HAVE_TDM 1
#endif

__device__ __forceinline__ unsigned lds_offset(const void* p) {
  return (unsigned)(unsigned long long)(AS3 void*)(void*)p;
}

#if defined(HAVE_TDM)
// 2D tile DMA: elements are 2 bytes. D# per ISA 8.3/8.4 (normal mode, 2D).
// 6-arg toolchain form: (g0, g1, g2, g3, g4, cpol).
__device__ __forceinline__ void tdm_load_2d(unsigned lds_off, const void* gaddr,
                                            unsigned tensor_d0, unsigned tensor_d1,
                                            unsigned tile_d0, unsigned tile_d1,
                                            unsigned stride0) {
  const unsigned long long ga = (unsigned long long)(size_t)gaddr;
  v4u g0;
  g0[0] = 1u;                                           // count=1, user mode
  g0[1] = lds_off;                                      // lds_addr (bytes)
  g0[2] = (unsigned)ga;                                 // global_addr[31:0]
  g0[3] = (unsigned)((ga >> 32) & 0x1FFFFFFull) | (2u << 30);  // [56:32]|type=2
  v8i g1;
  g1[0] = (int)(1u << 16);                              // data_size=1 (2 bytes)
  g1[1] = (int)((tensor_d0 & 0xFFFFu) << 16);           // tensor_dim0[15:0]
  g1[2] = (int)(((tensor_d0 >> 16) & 0xFFFFu) |
                ((tensor_d1 & 0xFFFFu) << 16));         // dim0[31:16]|dim1[15:0]
  g1[3] = (int)(((tensor_d1 >> 16) & 0xFFFFu) |
                (tile_d0 << 16));                       // dim1[31:16]|tile_dim0
  g1[4] = (int)(tile_d1 & 0xFFFFu);                     // tile_dim1, tile_dim2=0
  g1[5] = (int)stride0;                                 // tensor_dim0_stride lo
  g1[6] = 0;                                            // stride hi | dim1_stride
  g1[7] = 0;
  v4i g2 = {0, 0, 0, 0};
  v4i g3 = {0, 0, 0, 0};
  v8i g4 = {0, 0, 0, 0, 0, 0, 0, 0};
  __builtin_amdgcn_tensor_load_to_lds(g0, g1, g2, g3, g4, 0);
}

template <int N>
__device__ __forceinline__ void wait_tensor() {
#if __has_builtin(__builtin_amdgcn_s_wait_tensorcnt)
  __builtin_amdgcn_s_wait_tensorcnt(N);
#else
  asm volatile("s_wait_tensorcnt %0" ::"i"(N) : "memory");
#endif
}
#endif  // HAVE_TDM

__device__ __forceinline__ float hmax16(float v) {
#pragma unroll
  for (int off = 1; off < 16; off <<= 1)
    v = fmaxf(v, __shfl_xor(v, off, 32));
  return v;
}
__device__ __forceinline__ float hsum16(float v) {
#pragma unroll
  for (int off = 1; off < 16; off <<= 1)
    v += __shfl_xor(v, off, 32);
  return v;
}

// ---------------------------------------------------------------------------
// Kernel 0: convert W (f32, [k][n]) -> Wt (f16, [w][n][k]) once.
// ---------------------------------------------------------------------------
__global__ __launch_bounds__(256)
void wtrans_kernel(const float* __restrict__ Wq, const float* __restrict__ Wk,
                   const float* __restrict__ Wv, _Float16* __restrict__ wt) {
  const int id = blockIdx.x * 256 + threadIdx.x;   // 24576 threads
  const int w  = id >> 13;                         // 8192 per weight
  const int r  = id & 8191;
  const int n  = r >> 7;                           // 0..63
  const int k0 = (r & 127) * 8;                    // 0..1016
  const float* W = (w == 0) ? Wq : (w == 1) ? Wk : Wv;
  _Float16* dst = wt + ((size_t)w << 16) + (size_t)n * H_ + k0;
#pragma unroll
  for (int j = 0; j < 8; ++j)
    dst[j] = (_Float16)W[(size_t)(k0 + j) * D_ + n];
}

// ---------------------------------------------------------------------------
// Kernel 1: fused QKV projection. 4 waves/block, each wave a 16-row tile of x.
// f16 Wt chunks (3 x 64 x 32 = 12 KB) async-staged to LDS, double buffered,
// shared by all waves. q,k stored f16 [bs][64]; v transposed [b][d][s].
// ---------------------------------------------------------------------------
__global__ __launch_bounds__(128)
void qkv_proj_kernel(const float* __restrict__ x,
                     const _Float16* __restrict__ wt,
                     const float* __restrict__ bq, const float* __restrict__ bk,
                     const float* __restrict__ bv,
                     _Float16* __restrict__ qh, _Float16* __restrict__ kh,
                     _Float16* __restrict__ vt) {
  __shared__ __align__(16) _Float16 wlds[2][3][64 * 32];  // 2 x 12 KB

  const int tid  = threadIdx.x;
  const int lane = tid & 31;
  const int wid  = tid >> 5;
  const int row0 = (blockIdx.x * 4 + wid) * 16;    // flat row over B*S
  const int half = lane >> 4;
  const int rc   = lane & 15;
  const int srow = tid >> 1;                       // 0..63 staging row
  const int shb  = (tid & 1) << 4;                 // 0/16 halves

  const float* Bs[3] = {bq, bk, bv};

  v8f acc[3][4];
#pragma unroll
  for (int w = 0; w < 3; ++w)
#pragma unroll
    for (int nt = 0; nt < 4; ++nt)
#pragma unroll
      for (int e = 0; e < 8; ++e) acc[w][nt][e] = 0.0f;

  // stage chunk 0: 3 weights x 2 b128 per thread
#pragma unroll
  for (int w = 0; w < 3; ++w)
#pragma unroll
    for (int j = 0; j < 2; ++j)
      cp_async_b128(&wlds[0][w][srow * 32 + shb + j * 8],
                    wt + ((size_t)w << 16) + (size_t)srow * H_ + shb + j * 8);

  for (int kci = 0; kci < 32; ++kci) {
    const int kc = kci * 32;
    if (kci + 1 < 32) {
      const int nb = (kci + 1) & 1;
#pragma unroll
      for (int w = 0; w < 3; ++w)
#pragma unroll
        for (int j = 0; j < 2; ++j)
          cp_async_b128(&wlds[nb][w][srow * 32 + shb + j * 8],
                        wt + ((size_t)w << 16) + (size_t)srow * H_ +
                            (kc + 32) + shb + j * 8);
      wait_async<6>();
    } else {
      wait_async<0>();
    }
    __syncthreads();

    // A fragment: 16x32 chunk of x, four b128 loads + pack-convert
    Frag a;
    const float* xrow = x + (size_t)(row0 + rc) * H_ + kc;
    {
      const float4 f0 = *(const float4*)(xrow + 8 * half);
      const float4 f1 = *(const float4*)(xrow + 8 * half + 4);
      const float4 f2 = *(const float4*)(xrow + 16 + 8 * half);
      const float4 f3 = *(const float4*)(xrow + 16 + 8 * half + 4);
      a.h[0]  = (_Float16)f0.x; a.h[1]  = (_Float16)f0.y;
      a.h[2]  = (_Float16)f0.z; a.h[3]  = (_Float16)f0.w;
      a.h[4]  = (_Float16)f1.x; a.h[5]  = (_Float16)f1.y;
      a.h[6]  = (_Float16)f1.z; a.h[7]  = (_Float16)f1.w;
      a.h[8]  = (_Float16)f2.x; a.h[9]  = (_Float16)f2.y;
      a.h[10] = (_Float16)f2.z; a.h[11] = (_Float16)f2.w;
      a.h[12] = (_Float16)f3.x; a.h[13] = (_Float16)f3.y;
      a.h[14] = (_Float16)f3.z; a.h[15] = (_Float16)f3.w;
    }

    const int cur = kci & 1;
#pragma unroll
    for (int w = 0; w < 3; ++w) {
#pragma unroll
      for (int nt = 0; nt < 4; ++nt) {
        Frag bf;
        load_frag_h16(bf, &wlds[cur][w][(nt * 16 + rc) * 32], half);
        acc[w][nt] = __builtin_amdgcn_wmma_f32_16x16x32_f16(
            false, a.v, false, bf.v, (short)0, acc[w][nt], false, false);
      }
    }
    __syncthreads();
  }

  // bias add + f16 store
#pragma unroll
  for (int w = 0; w < 3; ++w) {
#pragma unroll
    for (int nt = 0; nt < 4; ++nt) {
      const int d = nt * 16 + rc;
      const float bias = Bs[w][d];
#pragma unroll
      for (int r = 0; r < 8; ++r) {
        const int row   = r + 8 * half;
        const int bsrow = row0 + row;
        const _Float16 hv = (_Float16)(acc[w][nt][r] + bias);
        if (w == 0) {
          qh[(size_t)bsrow * D_ + d] = hv;
        } else if (w == 1) {
          kh[(size_t)bsrow * D_ + d] = hv;
        } else {
          const int b = bsrow >> 11;               // / S_
          const int s = bsrow & (S_ - 1);
          vt[((size_t)b * D_ + d) * S_ + s] = hv;
        }
      }
    }
  }
}

// ---------------------------------------------------------------------------
// Kernel 2: flash attention. Block = 4 waves = 64 query rows of one batch.
// K tile (32x64 f16) and V^T tile (64x32 f16) staged into LDS by the Tensor
// Data Mover (TDM, TENSORcnt), double buffered; wave 0 issues K, wave 1
// issues V. Online softmax; P transposed C->A via per-wave LDS buffer.
// ---------------------------------------------------------------------------
__global__ __launch_bounds__(128)
void flash_attn_kernel(const _Float16* __restrict__ qh,
                       const _Float16* __restrict__ kh,
                       const _Float16* __restrict__ vt,
                       const int* __restrict__ mask,
                       float* __restrict__ out) {
  __shared__ __align__(16) _Float16 kbuf[2][32 * 64];  // 2 x 4 KB
  __shared__ __align__(16) _Float16 vbuf[2][64 * 32];  // 2 x 4 KB
  __shared__ __align__(16) _Float16 plds[4][16 * 32];  // 1 KB per wave

  const int tid  = threadIdx.x;
  const int lane = tid & 31;
  const int wid  = tid >> 5;
  const int b    = blockIdx.x >> 5;                    // 32 blocks per batch
  const int q0   = ((blockIdx.x & 31) << 6) + (wid << 4);
  const int half = lane >> 4;
  const int rc   = lane & 15;

  const _Float16* kbase = kh + (size_t)(b * S_) * D_;  // [key][64]
  const _Float16* vbase = vt + (size_t)(b * D_) * S_;  // [d][2048]

  // Q fragments (A-layout): two b128 loads each
  Frag a0, a1;
  {
    const _Float16* qrow = qh + (size_t)(b * S_ + q0 + rc) * D_;
    load_frag_h16(a0, qrow, half);
    load_frag_h16(a1, qrow + 32, half);
  }

  float m[8], l[8];
  v8f acc[4];
#pragma unroll
  for (int r = 0; r < 8; ++r) { m[r] = -1e30f; l[r] = 0.0f; }
#pragma unroll
  for (int nt = 0; nt < 4; ++nt)
#pragma unroll
    for (int e = 0; e < 8; ++e) acc[nt][e] = 0.0f;

  const float scale = 0.125f;                          // 1/sqrt(64)

  // ---- staging: TDM (preferred) or async-LDS fallback ----------------------
#if defined(HAVE_TDM)
  if (wid == 0)
    tdm_load_2d(lds_offset(&kbuf[0][0]), kbase, D_, S_, D_, 32, D_);
  else if (wid == 1)
    tdm_load_2d(lds_offset(&vbuf[0][0]), vbase, S_, D_, 32, D_, S_);
#else
  const int vrow = tid >> 1;
  const int vhb  = (tid & 1) << 4;
#pragma unroll
  for (int j = 0; j < 2; ++j) {
    cp_async_b128(&kbuf[0][tid * 16 + j * 8], kbase + tid * 16 + j * 8);
    cp_async_b128(&vbuf[0][vrow * 32 + vhb + j * 8],
                  vbase + (size_t)vrow * S_ + vhb + j * 8);
  }
#endif

  for (int kti = 0; kti < S_ / 32; ++kti) {
    const int kt = kti * 32;
    if (kti + 1 < S_ / 32) {
      const int nb = (kti + 1) & 1;
#if defined(HAVE_TDM)
      if (wid == 0)
        tdm_load_2d(lds_offset(&kbuf[nb][0]), kbase + (size_t)(kt + 32) * D_,
                    D_, S_, D_, 32, D_);
      else if (wid == 1)
        tdm_load_2d(lds_offset(&vbuf[nb][0]), vbase + kt + 32,
                    S_, D_, 32, D_, S_);
      wait_tensor<1>();
#else
#pragma unroll
      for (int j = 0; j < 2; ++j) {
        cp_async_b128(&kbuf[nb][tid * 16 + j * 8],
                      kbase + (size_t)(kt + 32) * D_ + tid * 16 + j * 8);
        cp_async_b128(&vbuf[nb][vrow * 32 + vhb + j * 8],
                      vbase + (size_t)vrow * S_ + kt + 32 + vhb + j * 8);
      }
      wait_async<4>();
#endif
    } else {
#if defined(HAVE_TDM)
      wait_tensor<0>();
#else
      wait_async<0>();
#endif
    }
    __syncthreads();

    const _Float16* kl = kbuf[kti & 1];
    const _Float16* vl = vbuf[kti & 1];

    // --- scores: S = Q @ K^T for 2 groups of 16 keys (ds_load_b128) --------
    Frag bk00, bk01, bk10, bk11;
    load_frag_h16(bk00, kl + rc * D_, half);
    load_frag_h16(bk01, kl + rc * D_ + 32, half);
    load_frag_h16(bk10, kl + (16 + rc) * D_, half);
    load_frag_h16(bk11, kl + (16 + rc) * D_ + 32, half);

    v8f s0, s1;
#pragma unroll
    for (int e = 0; e < 8; ++e) { s0[e] = 0.0f; s1[e] = 0.0f; }
    s0 = __builtin_amdgcn_wmma_f32_16x16x32_f16(false, a0.v, false, bk00.v,
                                                (short)0, s0, false, false);
    s0 = __builtin_amdgcn_wmma_f32_16x16x32_f16(false, a1.v, false, bk01.v,
                                                (short)0, s0, false, false);
    s1 = __builtin_amdgcn_wmma_f32_16x16x32_f16(false, a0.v, false, bk10.v,
                                                (short)0, s1, false, false);
    s1 = __builtin_amdgcn_wmma_f32_16x16x32_f16(false, a1.v, false, bk11.v,
                                                (short)0, s1, false, false);

    const int mk0 = mask[b * S_ + kt + rc];
    const int mk1 = mask[b * S_ + kt + 16 + rc];

    // --- online softmax update --------------------------------------------
    float p0[8], p1[8], alpha[8];
#pragma unroll
    for (int r = 0; r < 8; ++r) {
      float v0 = s0[r] * scale; if (mk0 == 0) v0 = -1e30f;
      float v1 = s1[r] * scale; if (mk1 == 0) v1 = -1e30f;
      const float rmax = hmax16(fmaxf(v0, v1));
      const float mn   = fmaxf(m[r], rmax);
      const float e0   = __expf(v0 - mn);
      const float e1   = __expf(v1 - mn);
      const float al   = __expf(m[r] - mn);
      const float rs   = hsum16(e0 + e1);
      l[r] = l[r] * al + rs;
      m[r] = mn;
      alpha[r] = al;
      p0[r] = e0;
      p1[r] = e1;
    }
#pragma unroll
    for (int nt = 0; nt < 4; ++nt)
#pragma unroll
      for (int r = 0; r < 8; ++r) acc[nt][r] *= alpha[r];

    // --- transpose P (C layout) -> A layout via per-wave LDS ---------------
    _Float16* pl = plds[wid];
#pragma unroll
    for (int r = 0; r < 8; ++r) {
      const int row = r + 8 * half;
      pl[row * 32 + rc]      = (_Float16)p0[r];
      pl[row * 32 + 16 + rc] = (_Float16)p1[r];
    }
    __syncthreads();
    Frag pa;
    load_frag_h16(pa, pl + rc * 32, half);

    // --- acc += P @ V (V^T fragments from LDS, ds_load_b128) ---------------
#pragma unroll
    for (int nt = 0; nt < 4; ++nt) {
      Frag bf;
      load_frag_h16(bf, vl + (nt * 16 + rc) * 32, half);
      acc[nt] = __builtin_amdgcn_wmma_f32_16x16x32_f16(
          false, pa.v, false, bf.v, (short)0, acc[nt], false, false);
    }
    __syncthreads();   // protect LDS P buffer and K/V buffers for reuse
  }

  // --- normalize and store -------------------------------------------------
#pragma unroll
  for (int nt = 0; nt < 4; ++nt) {
    const int d = nt * 16 + rc;
#pragma unroll
    for (int r = 0; r < 8; ++r) {
      const int row = r + 8 * half;
      out[(size_t)(b * S_ + q0 + row) * D_ + d] = acc[nt][r] / l[r];
    }
  }
}

// ---------------------------------------------------------------------------
extern "C" void kernel_launch(void* const* d_in, const int* in_sizes, int n_in,
                              void* d_out, int out_size, void* d_ws, size_t ws_size,
                              hipStream_t stream) {
  const float* x    = (const float*)d_in[0];
  const int*   mask = (const int*)d_in[1];
  const float* Wq   = (const float*)d_in[2];
  const float* bq   = (const float*)d_in[3];
  const float* Wk   = (const float*)d_in[4];
  const float* bk   = (const float*)d_in[5];
  const float* Wv   = (const float*)d_in[6];
  const float* bv   = (const float*)d_in[7];
  float* out = (float*)d_out;

  char* ws = (char*)d_ws;
  _Float16* qh = (_Float16*)ws;                                  // 2 MB
  _Float16* kh = (_Float16*)(ws + (size_t)2 * 1024 * 1024);      // 2 MB
  _Float16* vt = (_Float16*)(ws + (size_t)4 * 1024 * 1024);      // 2 MB
  _Float16* wt = (_Float16*)(ws + (size_t)6 * 1024 * 1024);      // 384 KB

  wtrans_kernel<<<96, 256, 0, stream>>>(Wq, Wk, Wv, wt);
  qkv_proj_kernel<<<256, 128, 0, stream>>>(x, wt, bq, bk, bv, qh, kh, vt);
  flash_attn_kernel<<<256, 128, 0, stream>>>(qh, kh, vt, mask, out);
}
